// HeteGAT_multi_84593675862337
// MI455X (gfx1250) — compile-verified, matrix operand order checked by reading, not run on hardware
//
#include <hip/hip_runtime.h>
#include <hip/hip_bf16.h>

// ---------------------------------------------------------------------------
// HeteGAT_multi for MI455X (gfx1250, wave32, WMMA).
// All three GEMM stages use v_wmma_f32_16x16x32_bf16 (f32 accumulate).
// Attention is flash-style: stats pass + fused recompute-P/WMMA pass; the
// [N,N,H] attention tensor is never materialized.
// ---------------------------------------------------------------------------

typedef __attribute__((ext_vector_type(16))) __bf16 v16bf;
typedef __attribute__((ext_vector_type(8)))  float  v8f;

#define WMMA_BF16(a, b, c) \
  __builtin_amdgcn_wmma_f32_16x16x32_bf16(false, (a), false, (b), (short)0, (c), false, false)

static constexpr int Mfull = 6000;   // full graph nodes
static constexpr int Ffull = 256;    // feature dim
static constexpr int Nn    = 3000;   // batch nodes (multiple of 8!)
static constexpr int Hh    = 8;      // heads
static constexpr int Oo    = 64;     // per-head out dim
static constexpr int Dd    = 512;    // H*O
static constexpr int Npad  = 3008;   // padded N (K-tile safety for sT reads)
static constexpr int NT16  = (Nn + 15) / 16;   // 188 16-row tiles
static constexpr int NT32  = (Nn + 31) / 32;   // 94  32-row tiles
static constexpr float NEG_SLOPE = 0.01f;

// K index inside a 32-wide K tile for the A/B fragments of
// v_wmma_*_16x16x32_bf16 (ISA 7.12.2, 16-bit A 16x32 layout), per lane/elem.
__device__ __forceinline__ int kmap(int lane, int e) {
  return ((lane & 16) ? 8 : 0) + e + ((e & 8) ? 8 : 0);
}

// Fragment load from a contiguous-K bf16 run (base points at K=0 of the
// 32-wide K tile). Two 16-byte (global_load_b128) loads per lane; base and
// row pitches are multiples of 16 B by construction.
__device__ __forceinline__ v16bf load_frag_bf16(const __bf16* base, int lane) {
  const int kb = (lane & 16) ? 8 : 0;
  const uint4* p = (const uint4*)(base + kb);
  union { uint4 u[2]; v16bf v; } u;
  u.u[0] = p[0];   // K = kb .. kb+7
  u.u[1] = p[2];   // K = kb+16 .. kb+23
  return u.v;
}

// Same, but from f32 memory (4 x float4 loads + pack converts).
__device__ __forceinline__ v16bf load_frag_f32(const float* base, int lane) {
  const int kb = (lane & 16) ? 8 : 0;
  const float4* p = (const float4*)(base + kb);
  const float4 x0 = p[0], x1 = p[1];   // K = kb .. kb+7
  const float4 y0 = p[4], y1 = p[5];   // K = kb+16 .. kb+23
  v16bf v;
  v[0] = (__bf16)x0.x; v[1] = (__bf16)x0.y; v[2]  = (__bf16)x0.z; v[3]  = (__bf16)x0.w;
  v[4] = (__bf16)x1.x; v[5] = (__bf16)x1.y; v[6]  = (__bf16)x1.z; v[7]  = (__bf16)x1.w;
  v[8] = (__bf16)y0.x; v[9] = (__bf16)y0.y; v[10] = (__bf16)y0.z; v[11] = (__bf16)y0.w;
  v[12] = (__bf16)y1.x; v[13] = (__bf16)y1.y; v[14] = (__bf16)y1.z; v[15] = (__bf16)y1.w;
  return v;
}

// ---------------------------------------------------------------------------
// Kernel 0: Wm [512,512] f32 -> WmT [col][k] bf16 (transposed for B-fragments)
// ---------------------------------------------------------------------------
__global__ __launch_bounds__(256)
void k_cvt_wm(const float* __restrict__ Wm, __bf16* __restrict__ WmT) {
  int idx = blockIdx.x * 256 + threadIdx.x;    // idx = d*512 + k
  if (idx >= Dd * Dd) return;
  int d = idx >> 9, k = idx & 511;
  WmT[idx] = (__bf16)Wm[k * Dd + d];
}

// ---------------------------------------------------------------------------
// Kernel 1: gathered per-head projection  s = x @ W + b   (K = Fin = 32 exact)
// One block = 8 waves = 8 heads, one 16-row tile. 4 WMMAs per wave.
// Writes s [N][H][O] f32 and sT [H][O][Npad] bf16 (transposed for k_aggr B).
// Each lane's 8 results are contiguous in i for sT -> one b128 store.
// ---------------------------------------------------------------------------
__global__ __launch_bounds__(256)
void k_proj(const float* __restrict__ feat, const int* __restrict__ nodes,
            const float* __restrict__ W, const float* __restrict__ bvec,
            float* __restrict__ s, __bf16* __restrict__ sT) {
  const int tid   = threadIdx.x;
  const int lane  = tid & 31;
  const int h     = tid >> 5;
  const int i0    = blockIdx.x * 16;
  const int row16 = lane & 15;
  const int half8 = (lane & 16) ? 8 : 0;
  const int cstart = i0 + half8;               // 8-row chunk start (Nn % 8 == 0)
  const bool cvalid = cstart < Nn;

  int ic = i0 + row16; if (ic >= Nn) ic = Nn - 1;
  const int node = nodes[ic];
  const v16bf a = load_frag_f32(feat + (size_t)node * Ffull + h * 32, lane);

  #pragma unroll
  for (int ot = 0; ot < 4; ++ot) {
    const int o = ot * 16 + row16;            // B column / C column for this lane
    v16bf b;
    #pragma unroll
    for (int e = 0; e < 16; ++e)
      b[e] = (__bf16)W[(h * 32 + kmap(lane, e)) * Oo + o];
    v8f c = {};
    c = WMMA_BF16(a, b, c);
    const float badd = bvec[h * Oo + o];

    float vv[8];
    union { __bf16 b8[8]; uint4 u; } pk;
    #pragma unroll
    for (int k = 0; k < 8; ++k) { vv[k] = c[k] + badd; pk.b8[k] = (__bf16)vv[k]; }
    if (cvalid) {
      *(uint4*)(sT + (size_t)(h * Oo + o) * Npad + cstart) = pk.u;   // b128 store
      #pragma unroll
      for (int k = 0; k < 8; ++k)
        s[((size_t)(cstart + k) * Hh + h) * Oo + o] = vv[k];
    }
  }
}

// ---------------------------------------------------------------------------
// Kernel 2: f1[n,h] = s[n,h,:]·a1[h,:] + a1b[h];  f2 likewise.
// ---------------------------------------------------------------------------
__global__ __launch_bounds__(256)
void k_f12(const float* __restrict__ s,
           const float* __restrict__ a1, const float* __restrict__ a1b,
           const float* __restrict__ a2, const float* __restrict__ a2b,
           float* __restrict__ f1, float* __restrict__ f2) {
  int idx = blockIdx.x * 256 + threadIdx.x;   // idx = n*8 + h
  if (idx >= Nn * Hh) return;
  const int h = idx & 7;
  const float* sp = s + (size_t)idx * Oo;
  float acc1 = a1b[h], acc2 = a2b[h];
  #pragma unroll 8
  for (int o = 0; o < Oo; ++o) {
    const float sv = sp[o];
    acc1 = fmaf(sv, a1[h * Oo + o], acc1);
    acc2 = fmaf(sv, a2[h * Oo + o], acc2);
  }
  f1[idx] = acc1;
  f2[idx] = acc2;
}

// ---------------------------------------------------------------------------
// Kernel 3: softmax stats (rowmax / rowsum) per (i, h), branch-free online
// update, one block per row i; each gathered bias element feeds all 8 heads.
// ---------------------------------------------------------------------------
__global__ __launch_bounds__(256)
void k_stats(const float* __restrict__ bias, const int* __restrict__ nodes,
             const float* __restrict__ f1, const float* __restrict__ f2,
             float* __restrict__ rowmax, float* __restrict__ rowsum) {
  __shared__ float smx[Hh][256];
  __shared__ float ssm[Hh][256];
  const int i   = blockIdx.x;
  const int tid = threadIdx.x;
  const float* brow = bias + (size_t)nodes[i] * Mfull;

  float f1h[Hh];
  #pragma unroll
  for (int h = 0; h < Hh; ++h) f1h[h] = f1[i * Hh + h];

  float mx[Hh], sm[Hh];
  #pragma unroll
  for (int h = 0; h < Hh; ++h) { mx[h] = -3.4e38f; sm[h] = 0.f; }

  for (int j = tid; j < Nn; j += 256) {
    if (j + 256 < Nn) __builtin_prefetch(&brow[nodes[j + 256]], 0, 1);
    const float bv = brow[nodes[j]];
    #pragma unroll
    for (int h = 0; h < Hh; ++h) {
      float l = f1h[h] + f2[j * Hh + h];
      l = (l >= 0.f) ? l : NEG_SLOPE * l;     // LeakyReLU before bias add
      l += bv;
      const float nm = fmaxf(mx[h], l);       // branch-free online update
      const float e  = __expf(fminf(mx[h], l) - nm);
      sm[h] = (l > mx[h]) ? fmaf(sm[h], e, 1.f) : (sm[h] + e);
      mx[h] = nm;
    }
  }
  #pragma unroll
  for (int h = 0; h < Hh; ++h) { smx[h][tid] = mx[h]; ssm[h][tid] = sm[h]; }
  __syncthreads();
  for (int off = 128; off > 0; off >>= 1) {
    if (tid < off) {
      #pragma unroll
      for (int h = 0; h < Hh; ++h) {
        const float m1 = smx[h][tid], m2 = smx[h][tid + off];
        const float s1 = ssm[h][tid], s2 = ssm[h][tid + off];
        const float m = fmaxf(m1, m2);
        ssm[h][tid] = s1 * __expf(m1 - m) + s2 * __expf(m2 - m);
        smx[h][tid] = m;
      }
    }
    __syncthreads();
  }
  if (tid < Hh) {
    rowmax[i * Hh + tid] = smx[tid][0];
    rowsum[i * Hh + tid] = ssm[tid][0];
  }
}

// ---------------------------------------------------------------------------
// Kernel 4: flash aggregation. One block = 32-row tile, 8 waves = 8 heads.
// Per K-step (32 j's): LDS-stage gathered 32x32 bias tile (shared by all 8
// heads), each wave builds TWO bf16 P fragments in registers and runs 8
// v_wmma accumulations against 4 shared B fragments from sT.
// ELU epilogue -> h1 bf16 [N][512].
// ---------------------------------------------------------------------------
__global__ __launch_bounds__(256)
void k_aggr(const float* __restrict__ bias, const int* __restrict__ nodes,
            const float* __restrict__ f1, const float* __restrict__ f2,
            const float* __restrict__ rowmax, const float* __restrict__ rowsum,
            const __bf16* __restrict__ sT, __bf16* __restrict__ h1bf) {
  __shared__ float  bb[32][33];     // padded vs bank conflicts
  __shared__ float  f2t[32][9];
  __shared__ size_t rowbase[32];

  const int tid   = threadIdx.x;
  const int lane  = tid & 31;
  const int h     = tid >> 5;
  const int i0    = blockIdx.x * 32;
  const int row16 = lane & 15;
  const int half8 = (lane & 16) ? 8 : 0;

  if (tid < 32) {
    int ii = i0 + tid; if (ii >= Nn) ii = Nn - 1;
    rowbase[tid] = (size_t)nodes[ii] * Mfull;
  }

  int icA = i0 + row16;      if (icA >= Nn) icA = Nn - 1;
  int icB = i0 + 16 + row16; if (icB >= Nn) icB = Nn - 1;
  const float fiA = f1[icA * Hh + h];
  const float rmA = rowmax[icA * Hh + h];
  const float riA = 1.f / rowsum[icA * Hh + h];
  const float fiB = f1[icB * Hh + h];
  const float rmB = rowmax[icB * Hh + h];
  const float riB = 1.f / rowsum[icB * Hh + h];

  v8f acc[2][4] = {};
  for (int jt = 0; jt < NT32; ++jt) {
    const int j0 = jt * 32;
    __syncthreads();                                   // WAR on LDS tiles
    #pragma unroll
    for (int t = tid; t < 1024; t += 256) {            // 32x32 bias tile
      const int ii = t >> 5, jj = t & 31;
      const int j  = j0 + jj;
      bb[ii][jj] = (j < Nn) ? bias[rowbase[ii] + (size_t)nodes[j]] : -1e30f;
    }
    {                                                  // 32x8 f2 tile
      const int jj = tid >> 3, hh = tid & 7;
      const int j  = j0 + jj;
      f2t[jj][hh] = (j < Nn) ? f2[j * Hh + hh] : 0.f;
    }
    __syncthreads();                                   // RAW on LDS tiles

    v16bf a0, a1;                                      // P fragments (bf16)
    #pragma unroll
    for (int e = 0; e < 16; ++e) {
      const int K  = kmap(lane, e);
      const float fz = f2t[K][h];
      float la = fiA + fz;
      la = (la >= 0.f) ? la : NEG_SLOPE * la;
      la += bb[row16][K];                              // -1e30 for masked j -> p = 0
      float lb = fiB + fz;
      lb = (lb >= 0.f) ? lb : NEG_SLOPE * lb;
      lb += bb[row16 + 16][K];
      a0[e] = (__bf16)(__expf(la - rmA) * riA);
      a1[e] = (__bf16)(__expf(lb - rmB) * riB);
    }

    #pragma unroll
    for (int ot = 0; ot < 4; ++ot) {
      const int o = ot * 16 + row16;
      const v16bf b = load_frag_bf16(sT + (size_t)(h * Oo + o) * Npad + j0, lane);
      acc[0][ot] = WMMA_BF16(a0, b, acc[0][ot]);
      acc[1][ot] = WMMA_BF16(a1, b, acc[1][ot]);
    }
  }

  // ELU epilogue; 8-row chunks are fully valid or fully invalid (Nn % 8 == 0)
  #pragma unroll
  for (int t = 0; t < 2; ++t) {
    const int cstart = i0 + t * 16 + half8;
    if (cstart < Nn) {
      #pragma unroll
      for (int ot = 0; ot < 4; ++ot) {
        #pragma unroll
        for (int k = 0; k < 8; ++k) {
          float v = acc[t][ot][k];
          v = (v > 0.f) ? v : (__expf(v) - 1.f);       // ELU (alpha = 1)
          h1bf[(size_t)(cstart + k) * Dd + h * Oo + ot * 16 + row16] = (__bf16)v;
        }
      }
    }
  }
}

// ---------------------------------------------------------------------------
// Kernel 5: h1 [N,512] @ Wm [512,512] + bm, fused rl scale, scattered directly
// into the transposed/concatenated output layout out[d*9000 + r*3000 + n].
// One block = 32-row tile; wave w owns cols [64w, 64w+64); B fragments are
// reused across the two 16-row A tiles. Each lane's 8 results are contiguous
// in n -> two b128 stores per tile.
// ---------------------------------------------------------------------------
__global__ __launch_bounds__(256)
void k_final(const __bf16* __restrict__ h1bf, const __bf16* __restrict__ WmT,
             const float* __restrict__ bm, const float* __restrict__ rl,
             float* __restrict__ out, int r) {
  const int tid   = threadIdx.x;
  const int lane  = tid & 31;
  const int w     = tid >> 5;
  const int i0    = blockIdx.x * 32;
  const int row16 = lane & 15;
  const int half8 = (lane & 16) ? 8 : 0;

  int icA = i0 + row16;      if (icA >= Nn) icA = Nn - 1;
  int icB = i0 + 16 + row16; if (icB >= Nn) icB = Nn - 1;
  const __bf16* apA = h1bf + (size_t)icA * Dd;
  const __bf16* apB = h1bf + (size_t)icB * Dd;

  v8f acc[2][4] = {};
  for (int kt = 0; kt < Dd / 32; ++kt) {               // 16 K-steps
    const int k0 = kt * 32;
    const v16bf a0 = load_frag_bf16(apA + k0, lane);
    const v16bf a1 = load_frag_bf16(apB + k0, lane);
    #pragma unroll
    for (int ot = 0; ot < 4; ++ot) {
      const int d = w * 64 + ot * 16 + row16;
      const v16bf b = load_frag_bf16(WmT + (size_t)d * Dd + k0, lane);
      acc[0][ot] = WMMA_BF16(a0, b, acc[0][ot]);
      acc[1][ot] = WMMA_BF16(a1, b, acc[1][ot]);
    }
  }

  #pragma unroll
  for (int t = 0; t < 2; ++t) {
    const int cstart = i0 + t * 16 + half8;            // 8-row chunk (Nn % 8 == 0)
    if (cstart < Nn) {
      const int c0 = r * Nn + cstart;
      const float4 r0 = *(const float4*)(rl + c0);
      const float4 r1 = *(const float4*)(rl + c0 + 4);
      #pragma unroll
      for (int ot = 0; ot < 4; ++ot) {
        const int d = w * 64 + ot * 16 + row16;
        const float bmd = bm[d];
        union { float f[8]; float4 q[2]; } pk;
        pk.f[0] = (acc[t][ot][0] + bmd) * r0.x;
        pk.f[1] = (acc[t][ot][1] + bmd) * r0.y;
        pk.f[2] = (acc[t][ot][2] + bmd) * r0.z;
        pk.f[3] = (acc[t][ot][3] + bmd) * r0.w;
        pk.f[4] = (acc[t][ot][4] + bmd) * r1.x;
        pk.f[5] = (acc[t][ot][5] + bmd) * r1.y;
        pk.f[6] = (acc[t][ot][6] + bmd) * r1.z;
        pk.f[7] = (acc[t][ot][7] + bmd) * r1.w;
        float4* op = (float4*)(out + (size_t)d * (3 * Nn) + c0);
        op[0] = pk.q[0];
        op[1] = pk.q[1];
      }
    }
  }
}

// ---------------------------------------------------------------------------
// Host launcher. setup_inputs() dict order:
//   [0..8]: (features_r, biases_r, batch_nodes_r) for r = 0,1,2
//   [9] W_attn [3,8,32,64]  [10] b_attn [3,8,64]
//   [11] a1 [3,8,64] [12] a1b [3,8] [13] a2 [3,8,64] [14] a2b [3,8]
//   [15] Wm [512,512] [16] bm [512] [17] rl [1, 9000]
// ---------------------------------------------------------------------------
extern "C" void kernel_launch(void* const* d_in, const int* in_sizes, int n_in,
                              void* d_out, int out_size, void* d_ws, size_t ws_size,
                              hipStream_t stream) {
  (void)in_sizes; (void)n_in; (void)out_size; (void)ws_size;

  char* ws = (char*)d_ws;
  auto aln = [](size_t x) { return (x + 255) & ~(size_t)255; };
  size_t off = 0;
  float*  s    = (float*)(ws + off);  off += aln((size_t)Nn * Hh * Oo * 4);
  __bf16* sT   = (__bf16*)(ws + off); off += aln((size_t)Hh * Oo * Npad * 2);
  float*  f1   = (float*)(ws + off);  off += aln((size_t)Nn * Hh * 4);
  float*  f2   = (float*)(ws + off);  off += aln((size_t)Nn * Hh * 4);
  float*  rmx  = (float*)(ws + off);  off += aln((size_t)Nn * Hh * 4);
  float*  rsm  = (float*)(ws + off);  off += aln((size_t)Nn * Hh * 4);
  __bf16* h1bf = (__bf16*)(ws + off); off += aln((size_t)Nn * Dd * 2);
  __bf16* WmT  = (__bf16*)(ws + off); off += aln((size_t)Dd * Dd * 2);

  const float* Wm = (const float*)d_in[15];
  const float* bm = (const float*)d_in[16];
  const float* rl = (const float*)d_in[17];
  float* out = (float*)d_out;

  k_cvt_wm<<<(Dd * Dd + 255) / 256, 256, 0, stream>>>(Wm, WmT);

  for (int r = 0; r < 3; ++r) {
    const float* feat  = (const float*)d_in[3 * r + 0];
    const float* bias  = (const float*)d_in[3 * r + 1];
    const int*   nodes = (const int*)d_in[3 * r + 2];
    const float* Wr  = (const float*)d_in[9]  + (size_t)r * Hh * 32 * Oo;
    const float* br  = (const float*)d_in[10] + (size_t)r * Hh * Oo;
    const float* a1  = (const float*)d_in[11] + (size_t)r * Hh * Oo;
    const float* a1b = (const float*)d_in[12] + (size_t)r * Hh;
    const float* a2  = (const float*)d_in[13] + (size_t)r * Hh * Oo;
    const float* a2b = (const float*)d_in[14] + (size_t)r * Hh;

    k_proj <<<NT16, 256, 0, stream>>>(feat, nodes, Wr, br, s, sT);
    k_f12  <<<(Nn * Hh + 255) / 256, 256, 0, stream>>>(s, a1, a1b, a2, a2b, f1, f2);
    k_stats<<<Nn, 256, 0, stream>>>(bias, nodes, f1, f2, rmx, rsm);
    k_aggr <<<NT32, 256, 0, stream>>>(bias, nodes, f1, f2, rmx, rsm, sT, h1bf);
    k_final<<<NT32, 256, 0, stream>>>(h1bf, WmT, bm, rl, out, r);
  }
}
